// DistanceLoss_47519518163176
// MI455X (gfx1250) — compile-verified, matrix operand order checked by reading, not run on hardware
//
#include <hip/hip_runtime.h>
#include <hip/hip_bf16.h>

typedef __attribute__((ext_vector_type(16))) _Float16 v16h;
typedef __attribute__((ext_vector_type(8)))  float    v8f;
typedef __attribute__((ext_vector_type(4)))  _Float16 v4h;

#define B_ROWS 4096
#define C_DIM  512
#define INV_T  10.0f

// ---------------------------------------------------------------------------
// Zero the accumulator regions of the workspace (rowsum + colsum).
// ---------------------------------------------------------------------------
__global__ void dl_zero_kernel(float* __restrict__ p, int n) {
    int i = blockIdx.x * 256 + threadIdx.x;
    if (i < n) p[i] = 0.0f;
}

// ---------------------------------------------------------------------------
// Row-wise L2 normalize (eps floor, as in reference) and convert to f16.
// grid = (4096, 2): y==0 -> batch1 -> p1h ; y==1 -> batch2 -> p2h.
// 128 threads, each handles 4 contiguous floats of the 512-wide row.
// ---------------------------------------------------------------------------
__global__ __launch_bounds__(128) void dl_normalize_kernel(
    const float* __restrict__ b1, const float* __restrict__ b2,
    _Float16* __restrict__ p1h, _Float16* __restrict__ p2h) {
    const int row = blockIdx.x;
    const float*  src = (blockIdx.y == 0) ? b1  : b2;
    _Float16*     dst = (blockIdx.y == 0) ? p1h : p2h;
    const int t = threadIdx.x;                       // 0..127
    const float4 v = *reinterpret_cast<const float4*>(src + (size_t)row * C_DIM + t * 4);
    float ss = v.x * v.x + v.y * v.y + v.z * v.z + v.w * v.w;
    // wave32 reduction
    for (int off = 16; off > 0; off >>= 1) ss += __shfl_xor(ss, off, 32);
    __shared__ float wsum[4];
    const int wv = t >> 5;
    if ((t & 31) == 0) wsum[wv] = ss;
    __syncthreads();
    const float total = wsum[0] + wsum[1] + wsum[2] + wsum[3];
    const float inv = 1.0f / fmaxf(sqrtf(total), 1e-8f);
    v4h h;
    h[0] = (_Float16)(v.x * inv);
    h[1] = (_Float16)(v.y * inv);
    h[2] = (_Float16)(v.z * inv);
    h[3] = (_Float16)(v.w * inv);
    *reinterpret_cast<v4h*>(dst + (size_t)row * C_DIM + t * 4) = h;
}

// ---------------------------------------------------------------------------
// Column sums of normalized p1 (for ||sum_i p1_i||^2 -> closed-form tril(pos)).
// 16 blocks x 512 threads; block b sums rows [256b, 256b+256), thread c = col.
// ---------------------------------------------------------------------------
__global__ __launch_bounds__(512) void dl_colsum_kernel(
    const _Float16* __restrict__ p1h, float* __restrict__ colsum) {
    const int c  = threadIdx.x;            // 0..511
    const int r0 = blockIdx.x * 256;
    float s = 0.0f;
    #pragma unroll 8
    for (int r = 0; r < 256; ++r)
        s += (float)p1h[(size_t)(r0 + r) * C_DIM + c];
    atomicAdd(&colsum[c], s);
}

// ---------------------------------------------------------------------------
// Fragment load: 16 f16 per lane as two B128 chunks, matching the CDNA5
// 16-bit 16x32 A/B operand layout (lanes 0-15: K{0..7,16..23}, lanes 16-31:
// K{8..15,24..31}) when the caller offsets the pointer by (lane>>4)*8.
// ---------------------------------------------------------------------------
__device__ __forceinline__ v16h dl_load_frag(const _Float16* p) {
    v16h f;
    float4* fp = reinterpret_cast<float4*>(&f);
    fp[0] = *reinterpret_cast<const float4*>(p);
    fp[1] = *reinterpret_cast<const float4*>(p + 16);
    return f;
}

// ---------------------------------------------------------------------------
// Fused GEMM (p1 @ p2^T, K=512, f16 WMMA, f32 acc) + exp(10*x) row-sum.
// Block = 128 threads (4 waves, 2x2). Block tile: 128x128.
// Wave tile: 64x64 = 4 A-frags x 4 B-frags -> 16 accumulators, 16 WMMAs per
// 8 fragment loads => 32 flops per L2 byte (2x round 1). Operands are fully
// L2-resident (8 MB << 192 MB), so no LDS staging.
// ---------------------------------------------------------------------------
__global__ __launch_bounds__(128) void dl_gemm_expsum_kernel(
    const _Float16* __restrict__ p1h, const _Float16* __restrict__ p2h,
    float* __restrict__ rowsum) {
    const int lane = threadIdx.x & 31;
    const int wv   = threadIdx.x >> 5;      // 0..3
    const int wm   = wv >> 1;               // 0..1
    const int wn   = wv & 1;                // 0..1
    const int M0   = blockIdx.y * 128 + wm * 64;
    const int N0   = blockIdx.x * 128 + wn * 64;
    const int hg   = lane >> 4;             // half-group: 0 or 1
    const int lr   = lane & 15;             // row/col within 16

    const _Float16* aBase = p1h + (size_t)(M0 + lr) * C_DIM + hg * 8;
    const _Float16* bBase = p2h + (size_t)(N0 + lr) * C_DIM + hg * 8;

    v8f acc[4][4] = {};

    for (int k = 0; k < C_DIM; k += 32) {
        v16h a[4], b[4];
        #pragma unroll
        for (int i = 0; i < 4; ++i) {
            a[i] = dl_load_frag(aBase + (size_t)(16 * i) * C_DIM + k);
            b[i] = dl_load_frag(bBase + (size_t)(16 * i) * C_DIM + k);
        }
        #pragma unroll
        for (int i = 0; i < 4; ++i) {
            #pragma unroll
            for (int j = 0; j < 4; ++j) {
                acc[i][j] = __builtin_amdgcn_wmma_f32_16x16x32_f16(
                    false, a[i], false, b[j], (short)0, acc[i][j], false, false);
            }
        }
    }

    // Epilogue: logits bounded by |cos|/T <= 10 -> direct exp is safe (no max
    // shift). C/D layout: VGPR r holds row M0+16i+r (lanes 0-15) / +8 more
    // (lanes 16-31); lane&15 = column. Fold the 4 n-tiles per m-tile, reduce
    // across 16-lane half-groups, one atomicAdd per row per wave.
    #pragma unroll
    for (int i = 0; i < 4; ++i) {
        #pragma unroll
        for (int r = 0; r < 8; ++r) {
            float e = 0.0f;
            #pragma unroll
            for (int j = 0; j < 4; ++j)
                e += __expf(acc[i][j][r] * INV_T);
            for (int off = 1; off < 16; off <<= 1)
                e += __shfl_xor(e, off, 32);
            if (lr == 0)
                atomicAdd(&rowsum[M0 + 16 * i + r + hg * 8], e);
        }
    }
}

// ---------------------------------------------------------------------------
// Finalize: loss = ( sum_i i*log(rowsum[i]) - (||s||^2 - B)/(2T) ) / nTerms.
// Single block, f64 accumulation for the 4096-term weighted sum.
// ---------------------------------------------------------------------------
__global__ __launch_bounds__(256) void dl_finalize_kernel(
    const float* __restrict__ rowsum, const float* __restrict__ colsum,
    float* __restrict__ out) {
    __shared__ double sd[256];
    const int t = threadIdx.x;

    float s2p = 0.0f;
    for (int c = t; c < C_DIM; c += 256) { const float v = colsum[c]; s2p += v * v; }
    sd[t] = (double)s2p;
    __syncthreads();
    for (int s = 128; s > 0; s >>= 1) { if (t < s) sd[t] += sd[t + s]; __syncthreads(); }
    const double s2 = sd[0];
    __syncthreads();

    double lp = 0.0;
    for (int i = t; i < B_ROWS; i += 256)
        lp += (double)i * (double)logf(rowsum[i]);
    sd[t] = lp;
    __syncthreads();
    for (int s = 128; s > 0; s >>= 1) { if (t < s) sd[t] += sd[t + s]; __syncthreads(); }

    if (t == 0) {
        const double Bd = (double)B_ROWS;
        const double nTerms = Bd * (Bd - 1.0) * 0.5;
        const double posTril = (s2 - Bd) / (2.0 * 0.1);   // sum of tril(pos), T=0.1
        out[0] = (float)((sd[0] - posTril) / nTerms);
    }
}

// ---------------------------------------------------------------------------
// Workspace layout (bytes):
//   [0,        4194304)  p1h  : 4096*512 f16
//   [4194304,  8388608)  p2h  : 4096*512 f16
//   [8388608,  8404992)  rowsum: 4096 f32
//   [8404992,  8407040)  colsum: 512 f32
// ---------------------------------------------------------------------------
extern "C" void kernel_launch(void* const* d_in, const int* in_sizes, int n_in,
                              void* d_out, int out_size, void* d_ws, size_t ws_size,
                              hipStream_t stream) {
    const float* b1 = (const float*)d_in[0];
    const float* b2 = (const float*)d_in[1];
    float* out = (float*)d_out;
    char*  ws  = (char*)d_ws;

    _Float16* p1h    = (_Float16*)(ws);
    _Float16* p2h    = (_Float16*)(ws + 4194304);
    float*    rowsum = (float*)(ws + 8388608);
    float*    colsum = (float*)(ws + 8388608 + 16384);

    // rowsum + colsum are contiguous: zero 4096 + 512 floats.
    dl_zero_kernel<<<(4608 + 255) / 256, 256, 0, stream>>>(rowsum, 4608);
    dl_normalize_kernel<<<dim3(B_ROWS, 2), 128, 0, stream>>>(b1, b2, p1h, p2h);
    dl_colsum_kernel<<<16, 512, 0, stream>>>(p1h, colsum);
    dl_gemm_expsum_kernel<<<dim3(B_ROWS / 128, B_ROWS / 128), 128, 0, stream>>>(p1h, p2h, rowsum);
    dl_finalize_kernel<<<1, 256, 0, stream>>>(rowsum, colsum, out);
}
